// CausalSelfAttention_26491358281789
// MI455X (gfx1250) — compile-verified, matrix operand order checked by reading, not run on hardware
//
#include <hip/hip_runtime.h>
#include <hip/hip_bf16.h>

#define B_SZ     2
#define S_TOK    2048
#define D_MODEL  1024
#define N_HEADS  16
#define HEAD_DIM 64

typedef __attribute__((ext_vector_type(16))) __bf16 v16bf;
typedef __attribute__((ext_vector_type(8)))  __bf16 v8bf;
typedef __attribute__((ext_vector_type(8)))  float  v8f;
typedef int v4i __attribute__((vector_size(16)));

// ---------------------------------------------------------------------------
// 16-byte global -> LDS copy: async (CDNA5 GLOBAL_LOAD_ASYNC_TO_LDS_B128) if
// the builtin exists, else register bounce.
// Probe-confirmed signature: (AS1 v4i*, AS3 v4i*, Imm offset, Imm cpol).
// ---------------------------------------------------------------------------
#if __has_builtin(__builtin_amdgcn_global_load_async_to_lds_b128)
#define HAS_ASYNC_LDS 1
#else
#define HAS_ASYNC_LDS 0
#endif

__device__ __forceinline__ void cp16(const __bf16* __restrict__ g,
                                     __bf16* __restrict__ l) {
#if HAS_ASYNC_LDS
    __builtin_amdgcn_global_load_async_to_lds_b128(
        (__attribute__((address_space(1))) v4i*)g,
        (__attribute__((address_space(3))) v4i*)l, 0, 0);
#else
    *(uint4*)l = *(const uint4*)g;
#endif
}

__device__ __forceinline__ void stage_fence() {
#if HAS_ASYNC_LDS
#if __has_builtin(__builtin_amdgcn_s_wait_asynccnt)
    __builtin_amdgcn_s_wait_asynccnt(0);
#else
    asm volatile("s_wait_asynccnt 0x0" ::: "memory");
#endif
#endif
    __syncthreads();
}

// ---------------------------------------------------------------------------
// Fragment load: `row` points at a 32-element K-contiguous run (64B aligned).
// Per ISA 7.12.2: lane-half 0 holds K 0..7 & 16..23, half 1 holds 8..15 & 24..31
// -> exactly two contiguous 16B chunks per lane (2x ds_load_b128).
// ---------------------------------------------------------------------------
__device__ __forceinline__ v16bf load_frag(const __bf16* row, int half) {
    v8bf lo = *(const v8bf*)(row + 8 * half);
    v8bf hi = *(const v8bf*)(row + 16 + 8 * half);
    return __builtin_shufflevector(lo, hi, 0, 1, 2, 3, 4, 5, 6, 7,
                                   8, 9, 10, 11, 12, 13, 14, 15);
}

// ---------------------------------------------------------------------------
// f32 -> bf16 convert (grid-stride)
// ---------------------------------------------------------------------------
__global__ void cvt_f32_bf16_kernel(const float* __restrict__ in,
                                    __bf16* __restrict__ out, size_t n) {
    size_t i = (size_t)blockIdx.x * blockDim.x + threadIdx.x;
    size_t stride = (size_t)gridDim.x * blockDim.x;
    for (; i < n; i += stride) out[i] = (__bf16)in[i];
}

// f32 [K][N] -> bf16 transposed [N][K]  (coalesced writes)
__global__ void cvt_transpose_bf16_kernel(const float* __restrict__ in,
                                          __bf16* __restrict__ out,
                                          int K, int N) {
    size_t idx = (size_t)blockIdx.x * blockDim.x + threadIdx.x;
    size_t total = (size_t)K * N;
    if (idx >= total) return;
    int k = (int)(idx % K);
    int n = (int)(idx / K);
    out[idx] = (__bf16)in[(size_t)k * N + n];
}

// ---------------------------------------------------------------------------
// bf16 GEMM: C[M][N] = A[M][K] @ Bt[N][K]^T + bias[N]  (f32 accumulate)
// One wave per 16x64 C tile. grid = (N/64, M/16), block = 32.
// Staging: straight-line async b128 copies; fragments: 2x ds_load_b128.
// ---------------------------------------------------------------------------
__global__ void gemm_bf16_wmma_kernel(const __bf16* __restrict__ A,
                                      const __bf16* __restrict__ Bt,
                                      const float* __restrict__ bias,
                                      float* __restrict__ C,
                                      int M, int N, int K) {
    __shared__ alignas(16) __bf16 As[16 * 32];   // [m][k]
    __shared__ alignas(16) __bf16 Bs[64 * 32];   // [n][k]

    const int lane = threadIdx.x;
    const int nl   = lane & 15;
    const int half = lane >> 4;
    const size_t m0 = (size_t)blockIdx.y * 16;
    const size_t n0 = (size_t)blockIdx.x * 64;

    v8f acc[4];
    for (int t = 0; t < 4; ++t) {
        float bv = bias[n0 + t * 16 + nl];
        for (int r = 0; r < 8; ++r) acc[t][r] = bv;
    }

    // per-lane fixed chunk coordinates (i = lane + it*32)
    for (int k0 = 0; k0 < K; k0 += 32) {
        const __bf16* Ag = A + m0 * (size_t)K + k0;
        const __bf16* Bg = Bt + n0 * (size_t)K + k0;
#pragma unroll
        for (int it = 0; it < 2; ++it) {         // A: 64 chunks of 16B
            int i = lane + it * 32;
            cp16(Ag + (i >> 2) * (size_t)K + (i & 3) * 8, As + i * 8);
        }
#pragma unroll
        for (int it = 0; it < 8; ++it) {         // B: 256 chunks of 16B
            int i = lane + it * 32;
            cp16(Bg + (i >> 2) * (size_t)K + (i & 3) * 8, Bs + i * 8);
        }
        stage_fence();

        v16bf a = load_frag(As + nl * 32, half);
#pragma unroll
        for (int t = 0; t < 4; ++t) {
            v16bf b = load_frag(Bs + (t * 16 + nl) * 32, half);
            acc[t] = __builtin_amdgcn_wmma_f32_16x16x32_bf16(
                false, a, false, b, (short)0, acc[t], false, false);
        }
        __syncthreads();
    }

#pragma unroll
    for (int t = 0; t < 4; ++t)
#pragma unroll
        for (int r = 0; r < 8; ++r) {
            int m = r + 8 * half;
            C[(m0 + m) * (size_t)N + n0 + t * 16 + nl] = acc[t][r];
        }
}

// ---------------------------------------------------------------------------
// RoPE + split qkv[B*S][3072] (f32):
//   q,k -> [B*H][S][64] bf16 (K-contiguous for A/B frags)
//   v   -> [B*H][64][S] bf16 (transposed: K-contiguous for P@V B frags)
// ---------------------------------------------------------------------------
__global__ void rope_split_kernel(const float* __restrict__ qkv,
                                  const float* __restrict__ cosT,
                                  const float* __restrict__ sinT,
                                  __bf16* __restrict__ Qo,
                                  __bf16* __restrict__ Ko,
                                  __bf16* __restrict__ Vo) {
    size_t idx = (size_t)blockIdx.x * blockDim.x + threadIdx.x;
    const size_t total = (size_t)B_SZ * N_HEADS * S_TOK * 32;
    if (idx >= total) return;

    int i = idx & 31;
    size_t rest = idx >> 5;
    int s = (int)(rest % S_TOK); rest /= S_TOK;
    int h = (int)(rest % N_HEADS);
    int b = (int)(rest / N_HEADS);
    int bh = b * N_HEADS + h;

    const float* base = qkv + ((size_t)b * S_TOK + s) * (3 * D_MODEL) + h * HEAD_DIM;
    float c  = cosT[s * 32 + i];
    float sn = sinT[s * 32 + i];

    float qe = base[2 * i],               qo = base[2 * i + 1];
    float ke = base[D_MODEL + 2 * i],     ko = base[D_MODEL + 2 * i + 1];
    float ve = base[2 * D_MODEL + 2 * i], vv = base[2 * D_MODEL + 2 * i + 1];

    size_t o = ((size_t)bh * S_TOK + s) * HEAD_DIM;
    Qo[o + 2 * i]     = (__bf16)(qe * c - qo * sn);
    Qo[o + 2 * i + 1] = (__bf16)(qe * sn + qo * c);
    Ko[o + 2 * i]     = (__bf16)(ke * c - ko * sn);
    Ko[o + 2 * i + 1] = (__bf16)(ke * sn + ko * c);

    size_t ov = (size_t)bh * HEAD_DIM;
    Vo[(ov + 2 * i) * S_TOK + s]     = (__bf16)ve;
    Vo[(ov + 2 * i + 1) * S_TOK + s] = (__bf16)vv;
}

// ---------------------------------------------------------------------------
// Flash attention: one wave per (bh, 16-row q tile), 32 keys/iteration.
// Q [bh][S][64], K [bh][S][64], V [bh][64][S] (transposed).
// ---------------------------------------------------------------------------
__global__ void attn_wmma_kernel(const __bf16* __restrict__ Q,
                                 const __bf16* __restrict__ Kd,
                                 const __bf16* __restrict__ Vd,
                                 __bf16* __restrict__ Out) {
    __shared__ alignas(16) __bf16 Qs[16 * 64];   // [qrow][hd]
    __shared__ alignas(16) __bf16 Ks[32 * 64];   // [key][hd]
    __shared__ alignas(16) __bf16 Vt[64 * 32];   // [hd][key]
    __shared__ alignas(16) __bf16 Ps[16 * 32];   // [qrow][key]

    const int lane = threadIdx.x;
    const int nl   = lane & 15;
    const int half = lane >> 4;
    const int bh = blockIdx.y;
    const int b  = bh / N_HEADS;
    const int h  = bh % N_HEADS;
    const int q0 = blockIdx.x * 16;

    const __bf16* Qp = Q + ((size_t)bh * S_TOK + q0) * HEAD_DIM;
#pragma unroll
    for (int it = 0; it < 4; ++it) {             // Q: 128 chunks of 16B
        int i = lane + it * 32;
        cp16(Qp + i * 8, Qs + i * 8);
    }
    stage_fence();

    v16bf qa0 = load_frag(Qs + nl * 64, half);
    v16bf qa1 = load_frag(Qs + nl * 64 + 32, half);

    float mrow[8], lrow[8];
    v8f o[4] = {};
    for (int r = 0; r < 8; ++r) { mrow[r] = -3.0e38f; lrow[r] = 0.0f; }

    const __bf16* Kbase = Kd + (size_t)bh * S_TOK * HEAD_DIM;
    const __bf16* Vbase = Vd + (size_t)bh * HEAD_DIM * S_TOK;

    const int kmax = q0 + 16;
    for (int kb = 0; kb * 32 < kmax; ++kb) {
        const int kbase = kb * 32;
        const __bf16* Kp = Kbase + (size_t)kbase * HEAD_DIM;   // 4KB contiguous
        const __bf16* Vp = Vbase + kbase;                      // 64B rows, stride S
#pragma unroll
        for (int it = 0; it < 8; ++it) {
            int i = lane + it * 32;
            cp16(Kp + i * 8, Ks + i * 8);
        }
#pragma unroll
        for (int it = 0; it < 8; ++it) {
            int i = lane + it * 32;                            // hd = i>>2, chunk = i&3
            cp16(Vp + (size_t)(i >> 2) * S_TOK + (i & 3) * 8, Vt + i * 8);
        }
        stage_fence();

        // S tile 16x32 = Q (16x64) @ K-block^T
        v8f s[2] = {};
#pragma unroll
        for (int t = 0; t < 2; ++t) {
            v16bf b0 = load_frag(Ks + (t * 16 + nl) * 64, half);
            v16bf b1 = load_frag(Ks + (t * 16 + nl) * 64 + 32, half);
            s[t] = __builtin_amdgcn_wmma_f32_16x16x32_bf16(
                false, qa0, false, b0, (short)0, s[t], false, false);
            s[t] = __builtin_amdgcn_wmma_f32_16x16x32_bf16(
                false, qa1, false, b1, (short)0, s[t], false, false);
        }

        // scale, causal mask, online softmax
        const float scale = 0.125f;
#pragma unroll
        for (int r = 0; r < 8; ++r) {
            const int qrow = q0 + r + 8 * half;
            for (int t = 0; t < 2; ++t) {
                int key = kbase + t * 16 + nl;
                float v = s[t][r] * scale;
                s[t][r] = (key <= qrow) ? v : -3.0e38f;
            }
            float mx = fmaxf(s[0][r], s[1][r]);
            for (int off = 1; off < 16; off <<= 1)
                mx = fmaxf(mx, __shfl_xor(mx, off, 16));
            float mnew = fmaxf(mrow[r], mx);
            float alpha = __expf(mrow[r] - mnew);
            float p0 = __expf(s[0][r] - mnew);
            float p1 = __expf(s[1][r] - mnew);
            s[0][r] = p0; s[1][r] = p1;
            float rs = p0 + p1;
            for (int off = 1; off < 16; off <<= 1)
                rs += __shfl_xor(rs, off, 16);
            lrow[r] = lrow[r] * alpha + rs;
            mrow[r] = mnew;
            for (int t = 0; t < 4; ++t) o[t][r] *= alpha;
        }

        // P: C-frag -> LDS [qrow][key] -> A-frag
#pragma unroll
        for (int t = 0; t < 2; ++t)
#pragma unroll
            for (int r = 0; r < 8; ++r)
                Ps[(r + 8 * half) * 32 + t * 16 + nl] = (__bf16)s[t][r];
        __syncthreads();

        v16bf pa = load_frag(Ps + nl * 32, half);
#pragma unroll
        for (int t = 0; t < 4; ++t) {
            v16bf vb = load_frag(Vt + (t * 16 + nl) * 32, half);
            o[t] = __builtin_amdgcn_wmma_f32_16x16x32_bf16(
                false, pa, false, vb, (short)0, o[t], false, false);
        }
        __syncthreads();
    }

#pragma unroll
    for (int t = 0; t < 4; ++t)
#pragma unroll
        for (int r = 0; r < 8; ++r) {
            int m = r + 8 * half;
            float val = o[t][r] / lrow[r];
            size_t tok = (size_t)q0 + m;
            Out[((size_t)b * S_TOK + tok) * D_MODEL + h * HEAD_DIM + t * 16 + nl] =
                (__bf16)val;
        }
}

// ---------------------------------------------------------------------------
// Host-side launcher
// ---------------------------------------------------------------------------
extern "C" void kernel_launch(void* const* d_in, const int* in_sizes, int n_in,
                              void* d_out, int out_size, void* d_ws, size_t ws_size,
                              hipStream_t stream) {
    const float* x        = (const float*)d_in[0];
    const float* rope_cos = (const float*)d_in[1];
    const float* rope_sin = (const float*)d_in[2];
    const float* Wqkv     = (const float*)d_in[3];
    const float* bqkv     = (const float*)d_in[4];
    const float* Wout     = (const float*)d_in[5];
    const float* bout     = (const float*)d_in[6];
    float* out            = (float*)d_out;

    const size_t nX    = (size_t)B_SZ * S_TOK * D_MODEL;
    const size_t nWqkv = (size_t)D_MODEL * 3 * D_MODEL;
    const size_t nWout = (size_t)D_MODEL * D_MODEL;
    const size_t nHead = (size_t)B_SZ * N_HEADS * S_TOK * HEAD_DIM;

    char* ws = (char*)d_ws;
    size_t off = 0;
    __bf16* xw      = (__bf16*)(ws + off); off += nX * 2;
    __bf16* wqkvt   = (__bf16*)(ws + off); off += nWqkv * 2;   // [3072][1024]
    __bf16* woutt   = (__bf16*)(ws + off); off += nWout * 2;   // [1024][1024]
    float*  qkv     = (float*)(ws + off);  off += (size_t)B_SZ * S_TOK * 3 * D_MODEL * 4;
    __bf16* qw      = (__bf16*)(ws + off); off += nHead * 2;
    __bf16* kw      = (__bf16*)(ws + off); off += nHead * 2;
    __bf16* vw      = (__bf16*)(ws + off); off += nHead * 2;   // transposed per head
    __bf16* attnout = (__bf16*)(ws + off); off += nX * 2;
    (void)ws_size; (void)in_sizes; (void)n_in; (void)out_size;

    cvt_f32_bf16_kernel<<<4096, 256, 0, stream>>>(x, xw, nX);
    cvt_transpose_bf16_kernel<<<(unsigned)((nWqkv + 255) / 256), 256, 0, stream>>>(
        Wqkv, wqkvt, D_MODEL, 3 * D_MODEL);
    cvt_transpose_bf16_kernel<<<(unsigned)((nWout + 255) / 256), 256, 0, stream>>>(
        Wout, woutt, D_MODEL, D_MODEL);

    {   // QKV projection
        dim3 grid(3 * D_MODEL / 64, B_SZ * S_TOK / 16);
        gemm_bf16_wmma_kernel<<<grid, 32, 0, stream>>>(
            xw, wqkvt, bqkv, qkv, B_SZ * S_TOK, 3 * D_MODEL, D_MODEL);
    }
    {   // RoPE + head split (+ V transpose)
        size_t total = (size_t)B_SZ * N_HEADS * S_TOK * 32;
        rope_split_kernel<<<(unsigned)((total + 255) / 256), 256, 0, stream>>>(
            qkv, rope_cos, rope_sin, qw, kw, vw);
    }
    {   // causal flash attention
        dim3 grid(S_TOK / 16, B_SZ * N_HEADS);
        attn_wmma_kernel<<<grid, 32, 0, stream>>>(qw, kw, vw, attnout);
    }
    {   // output projection
        dim3 grid(D_MODEL / 64, B_SZ * S_TOK / 16);
        gemm_bf16_wmma_kernel<<<grid, 32, 0, stream>>>(
            attnout, woutt, bout, out, B_SZ * S_TOK, D_MODEL, D_MODEL);
    }
}
